// dnbRNN_13365938225208
// MI455X (gfx1250) — compile-verified, matrix operand order checked by reading, not run on hardware
//
#include <hip/hip_runtime.h>
#include <hip/hip_bf16.h>

#define NU 64
#define NX 768
#define NY 32
#define BB 64
#define TT 1024
#define NBLK 12                 // persistent blocks for the recurrence
#define PITCH_NX (NX + 8)       // 776 (bank-conflict padding)
#define PITCH_NU (NU + 8)       // 72
#define USE_ASYNC_STAGE 1       // gfx1250 GLOBAL_LOAD_ASYNC_TO_LDS path

typedef __attribute__((ext_vector_type(16))) __bf16 v16bf;
typedef __attribute__((ext_vector_type(8)))  float  v8f;

union FragBF { v16bf v; unsigned u[8]; };
union FragF  { v8f  v; float    f[8]; };

__device__ __forceinline__ unsigned short f2bf(float x) {
  union { float f; unsigned u; } a; a.f = x;
  unsigned r = a.u + 0x7FFFu + ((a.u >> 16) & 1u);   // round-to-nearest-even
  return (unsigned short)(r >> 16);
}
__device__ __forceinline__ float bf2f(unsigned short s) {
  union { unsigned u; float f; } a; a.u = ((unsigned)s) << 16;
  return a.f;
}

// Load a 16x32 bf16 A/B fragment from a (padded) row-major ushort array.
// ISA layout: lane&15 = row; lanes>=16 shift K by +8; VGPR0-3: K pairs at kb,
// VGPR4-7: K pairs at kb+16.
__device__ __forceinline__ v16bf load_frag(const unsigned short* base, int pitch,
                                           int rowBase, int kBase) {
  const int lane = threadIdx.x & 31;
  const int rb = (rowBase + (lane & 15)) * pitch;
  const int kb = kBase + ((lane >> 4) << 3);
  const unsigned* p = (const unsigned*)base;   // pitch & kBase are even -> aligned
  FragBF f;
#pragma unroll
  for (int j = 0; j < 4; ++j) f.u[j]     = p[(rb + kb + 2 * j) >> 1];
#pragma unroll
  for (int j = 0; j < 4; ++j) f.u[4 + j] = p[(rb + kb + 16 + 2 * j) >> 1];
  return f.v;
}

__device__ __forceinline__ v8f wmma_bf16(v16bf a, v16bf b, v8f c) {
  return __builtin_amdgcn_wmma_f32_16x16x32_bf16(false, a, false, b,
                                                 (short)0, c, false, false);
}

// ---------------------------------------------------------------------------
// Kernel 0: states[0] = bf16(h0); reset grid barrier (every launch).
// ---------------------------------------------------------------------------
__global__ void init_kernel(const float* __restrict__ h0,
                            unsigned short* __restrict__ Sbuf,
                            unsigned* __restrict__ bar) {
  int i = blockIdx.x * blockDim.x + threadIdx.x;
  if (i < BB * NX) Sbuf[i] = f2bf(h0[i]);
  if (i == 0) *bar = 0u;
}

// ---------------------------------------------------------------------------
// Kernel 1: G[t][b][n] = Hb[n] + sum_k Kw[n][k] * u[b][k][t]   (WMMA, bf16)
// grid = B * (T/64) * 4 n-groups; block = 256 (8 waves)
// ---------------------------------------------------------------------------
__global__ void __launch_bounds__(256)
gproj_kernel(const float* __restrict__ u, const float* __restrict__ Kw,
             const float* __restrict__ Hb, unsigned short* __restrict__ Gbuf) {
  extern __shared__ unsigned short smem[];
  unsigned short* Ks = smem;                   // 192 x PITCH_NU  (n rows)
  unsigned short* Us = smem + 192 * PITCH_NU;  // 64  x PITCH_NU  (t rows)

  const int ng = blockIdx.x & 3;
  const int tc = (blockIdx.x >> 2) & 15;
  const int b  = blockIdx.x >> 6;
  const int t0 = tc * 64;
  const int n0 = ng * 192;
  const int tid = threadIdx.x;

  for (int idx = tid; idx < 192 * NU; idx += 256) {
    int n = idx / NU, k = idx % NU;
    Ks[n * PITCH_NU + k] = f2bf(Kw[(n0 + n) * NU + k]);
  }
  for (int idx = tid; idx < 64 * NU; idx += 256) {
    int k = idx >> 6, t = idx & 63;            // t fastest -> coalesced global
    Us[t * PITCH_NU + k] = f2bf(u[((size_t)b * NU + k) * TT + t0 + t]);
  }
  __syncthreads();

  const int w = tid >> 5, lane = tid & 31;
  const int mt = w & 3;                        // t-tile
  const int nt0 = (w >> 2) * 6;                // 6 n-tiles per wave

  FragF c[6];
#pragma unroll
  for (int i = 0; i < 6; ++i) {
    float hb = Hb[n0 + (nt0 + i) * 16 + (lane & 15)];
#pragma unroll
    for (int j = 0; j < 8; ++j) c[i].f[j] = hb;
  }
#pragma unroll
  for (int kc = 0; kc < NU / 32; ++kc) {
    v16bf a = load_frag(Us, PITCH_NU, mt * 16, kc * 32);
#pragma unroll
    for (int i = 0; i < 6; ++i) {
      v16bf bm = load_frag(Ks, PITCH_NU, (nt0 + i) * 16, kc * 32);
      c[i].v = wmma_bf16(a, bm, c[i].v);
    }
  }
#pragma unroll
  for (int i = 0; i < 6; ++i) {
    int n = n0 + (nt0 + i) * 16 + (lane & 15);
#pragma unroll
    for (int j = 0; j < 8; ++j) {
      int t = t0 + mt * 16 + j + 8 * (lane >> 4);
      Gbuf[((size_t)t * BB + b) * NX + n] = f2bf(c[i].f[j]);
    }
  }
}

// ---------------------------------------------------------------------------
// Kernel 2: persistent recurrence. 12 blocks, each owns a 64-col slice of Hw.
// Per step: async-copy h_t (bf16, 96KB) global->LDS, 48 v_wmma per wave,
// relu, write bf16 slice of h_{t+1}, device-scope atomic barrier.
// ---------------------------------------------------------------------------
__global__ void __launch_bounds__(256)
rnn_kernel(const float* __restrict__ Hw, const unsigned short* __restrict__ Gbuf,
           unsigned short* __restrict__ Sbuf, unsigned* __restrict__ bar) {
  extern __shared__ unsigned short smem[];
  unsigned short* Wl = smem;                   // 64 x PITCH_NX (n rows of Hw slice)
  unsigned short* Hs = smem + 64 * PITCH_NX;   // 64 x PITCH_NX (b rows of h_t)
  const unsigned HS_BYTE_BASE = 64u * PITCH_NX * 2u;  // dynamic-LDS offset of Hs

  const int nb  = blockIdx.x;
  const int tid = threadIdx.x;
  const int n0  = nb * (NX / NBLK);

  // one-time: Hw slice -> LDS (bf16), coalesced over k
  for (int idx = tid; idx < 64 * NX; idx += 256) {
    int n = idx / NX, k = idx % NX;
    Wl[n * PITCH_NX + k] = f2bf(Hw[(size_t)(n0 + n) * NX + k]);
  }

  const int w = tid >> 5, lane = tid & 31;
  const int mt  = w & 3;                       // b-tile
  const int nt0 = (w >> 2) * 2;                // 2 n-tiles per wave

  for (int s = 0; s < TT - 1; ++s) {
    const unsigned short* Sp = Sbuf + (size_t)s * BB * NX;
    // stage h_t: 6144 x 16B, lane-parallel
    for (int idx = tid; idx < (BB * NX) / 8; idx += 256) {
      int e = idx * 8;
      int r = e / NX, col = e % NX;            // col multiple of 8 -> 16B aligned
      unsigned ldsOff = HS_BYTE_BASE + (unsigned)(r * PITCH_NX + col) * 2u;
#if USE_ASYNC_STAGE
      unsigned long long ga = (unsigned long long)(const void*)(Sp + e);
      asm volatile("global_load_async_to_lds_b128 %0, %1, off"
                   :: "v"(ldsOff), "v"(ga) : "memory");
#else
      *(uint4*)&Hs[r * PITCH_NX + col] = *(const uint4*)&Sp[e];
#endif
    }

    // C-init from G[s] while the async copy is in flight
    FragF c[2];
#pragma unroll
    for (int i = 0; i < 2; ++i) {
      int n = n0 + (nt0 + i) * 16 + (lane & 15);
#pragma unroll
      for (int j = 0; j < 8; ++j) {
        int m = mt * 16 + j + 8 * (lane >> 4);
        c[i].f[j] = bf2f(Gbuf[((size_t)s * BB + m) * NX + n]);
      }
    }
    // prefetch next step's G slice (global_prefetch_b8)
    if (s + 1 < TT - 1 && tid < BB) {
      __builtin_prefetch(&Gbuf[((size_t)(s + 1) * BB + tid) * NX + n0], 0, 1);
    }

#if USE_ASYNC_STAGE
    asm volatile("s_wait_asynccnt 0x0" ::: "memory");
#endif
    __syncthreads();

#pragma unroll 4
    for (int kc = 0; kc < NX / 32; ++kc) {
      v16bf a = load_frag(Hs, PITCH_NX, mt * 16, kc * 32);
#pragma unroll
      for (int i = 0; i < 2; ++i) {
        v16bf bm = load_frag(Wl, PITCH_NX, (nt0 + i) * 16, kc * 32);
        c[i].v = wmma_bf16(a, bm, c[i].v);
      }
    }
    unsigned short* So = Sbuf + (size_t)(s + 1) * BB * NX;
#pragma unroll
    for (int i = 0; i < 2; ++i) {
      int n = n0 + (nt0 + i) * 16 + (lane & 15);
#pragma unroll
      for (int j = 0; j < 8; ++j) {
        int m = mt * 16 + j + 8 * (lane >> 4);
        So[(size_t)m * NX + n] = f2bf(fmaxf(c[i].f[j], 0.0f));
      }
    }
    // grid barrier: release our h_{t+1} slice, acquire everyone else's
    __threadfence();
    __syncthreads();
    if (tid == 0) {
      __hip_atomic_fetch_add(bar, 1u, __ATOMIC_RELEASE, __HIP_MEMORY_SCOPE_AGENT);
      const unsigned target = (unsigned)NBLK * (unsigned)(s + 1);
      while (__hip_atomic_load(bar, __ATOMIC_ACQUIRE, __HIP_MEMORY_SCOPE_AGENT) < target) {
        __builtin_amdgcn_s_sleep(2);
      }
    }
    __syncthreads();
  }
}

// ---------------------------------------------------------------------------
// Kernel 3: y[b][o][t] = Ob[o] + sum_n states[t][b][n] * Ow[o][n]  (WMMA)
// grid = 128 blocks x 8 t each; Ow staged in LDS, A read from global bf16.
// ---------------------------------------------------------------------------
__global__ void __launch_bounds__(256)
oproj_kernel(const unsigned short* __restrict__ Sbuf, const float* __restrict__ Ow,
             const float* __restrict__ Ob, float* __restrict__ out) {
  extern __shared__ unsigned short smem[];
  unsigned short* Os = smem;                   // NY x PITCH_NX
  const int tid = threadIdx.x;
  for (int idx = tid; idx < NY * NX; idx += 256) {
    int o = idx / NX, k = idx % NX;
    Os[o * PITCH_NX + k] = f2bf(Ow[o * NX + k]);
  }
  __syncthreads();

  const int w = tid >> 5, lane = tid & 31;
  const int mt = w & 3;                        // b-tile
  const int nt = w >> 2;                       // o-tile (0..1)
  const int tPer = TT / gridDim.x;

  for (int ti = 0; ti < tPer; ++ti) {
    int t = blockIdx.x * tPer + ti;
    const unsigned short* Sp = Sbuf + (size_t)t * BB * NX;
    FragF c;
    float ob = Ob[nt * 16 + (lane & 15)];
#pragma unroll
    for (int j = 0; j < 8; ++j) c.f[j] = ob;
#pragma unroll 4
    for (int kc = 0; kc < NX / 32; ++kc) {
      v16bf a  = load_frag(Sp, NX, mt * 16, kc * 32);   // global, unpadded pitch
      v16bf bm = load_frag(Os, PITCH_NX, nt * 16, kc * 32);
      c.v = wmma_bf16(a, bm, c.v);
    }
    int o = nt * 16 + (lane & 15);
#pragma unroll
    for (int j = 0; j < 8; ++j) {
      int b = mt * 16 + j + 8 * (lane >> 4);
      out[((size_t)b * NY + o) * TT + t] = c.f[j];
    }
  }
}

// ---------------------------------------------------------------------------
extern "C" void kernel_launch(void* const* d_in, const int* in_sizes, int n_in,
                              void* d_out, int out_size, void* d_ws, size_t ws_size,
                              hipStream_t stream) {
  const float* u  = (const float*)d_in[0];   // (B, NU, T)
  const float* h0 = (const float*)d_in[1];   // (B, NX)
  const float* Hw = (const float*)d_in[2];   // (1, NX, NX)
  const float* Hb = (const float*)d_in[3];   // (1, NX)
  const float* Kw = (const float*)d_in[4];   // (1, NX, NU)
  const float* Ow = (const float*)d_in[5];   // (NY, NX)
  const float* Ob = (const float*)d_in[6];   // (NY,)
  float* out = (float*)d_out;                // (B, NY, T) fp32

  const size_t SB = (size_t)TT * BB * NX;    // elements per bf16 state/G buffer
  unsigned short* Sbuf = (unsigned short*)d_ws;  // states, bf16, [t][b][n]
  unsigned short* Gbuf = Sbuf + SB;              // Kw*u_t + Hb, bf16, [t][b][n]
  unsigned* bar = (unsigned*)(Gbuf + SB);        // grid barrier counter

  init_kernel<<<(BB * NX) / 256, 256, 0, stream>>>(h0, Sbuf, bar);
  gproj_kernel<<<BB * (TT / 64) * 4, 256,
                 (192 + 64) * PITCH_NU * sizeof(unsigned short), stream>>>(u, Kw, Hb, Gbuf);
  rnn_kernel<<<NBLK, 256,
               2 * 64 * PITCH_NX * sizeof(unsigned short), stream>>>(Hw, Gbuf, Sbuf, bar);
  oproj_kernel<<<128, 256,
                 NY * PITCH_NX * sizeof(unsigned short), stream>>>(Sbuf, Ow, Ob, out);
}
